// QuantumAwareEncoder_67800353734918
// MI455X (gfx1250) — compile-verified, matrix operand order checked by reading, not run on hardware
//
#include <hip/hip_runtime.h>
#include <hip/hip_bf16.h>
#include <math.h>

typedef __attribute__((ext_vector_type(16))) _Float16 v16h;
typedef __attribute__((ext_vector_type(8)))  _Float16 v8h;
typedef __attribute__((ext_vector_type(4)))  _Float16 v4h;
typedef __attribute__((ext_vector_type(8)))  float    v8f;

#define HID 256
#define NHEAD 8
#define HDIM 32
#define NLAYER 5
#define FIN 34

// ---------------- helpers ----------------
__device__ __forceinline__ unsigned f2ord(float f) {
    unsigned b = __float_as_uint(f);
    return (b & 0x80000000u) ? ~b : (b | 0x80000000u);
}
__device__ __forceinline__ float ord2f(unsigned u) {
    unsigned b = (u & 0x80000000u) ? (u ^ 0x80000000u) : ~u;
    return __uint_as_float(b);
}
#define ORD_NEG_INF 0x007FFFFFu   // f2ord(-inf)

__device__ __forceinline__ float gelu_exact(float x) {
    return 0.5f * x * (1.0f + erff(x * 0.7071067811865476f));
}

// ---------------- fill ----------------
__global__ void fill_u32_kernel(unsigned* __restrict__ p, unsigned v, long long n) {
    long long i = (long long)blockIdx.x * 256 + threadIdx.x;
    if (i < n) p[i] = v;
}

// ---------------- pad x: [N,FIN] f32 -> [N,64] f32 zero-padded ----------------
__global__ void pad_x_kernel(const float* __restrict__ x, float* __restrict__ xp, int N) {
    long long i = (long long)blockIdx.x * 256 + threadIdx.x;
    if (i >= (long long)N * 64) return;
    int n = (int)(i >> 6), k = (int)(i & 63);
    xp[i] = (k < FIN) ? x[(size_t)n * FIN + k] : 0.0f;
}

// ---------------- weight prep: W[K][Nout] f32 -> Wt[Nout][Kpad] f16 (zero-padded K) ----------------
__global__ void prep_wt_kernel(const float* __restrict__ W, _Float16* __restrict__ Wt,
                               int K, int Nout, int Kpad) {
    int i = blockIdx.x * 256 + threadIdx.x;
    if (i >= Nout * Kpad) return;
    int n = i / Kpad, k = i - n * Kpad;
    Wt[i] = (k < K) ? (_Float16)W[(size_t)k * Nout + n] : (_Float16)0.0f;
}

// ---------------- WMMA GEMM: C[M,Nout] = A[M,KPAD](f32) @ Wt[Nout,KPAD](f16,B^T) + bias ----
// block = 256 threads = 8 waves; block tile = 16(M) x 128(N); wave tile = 16x16; K-step 32.
// lda == KPAD (A pre-padded); K loop fully unrolled (KPAD compile-time).
template <int KPAD>
__global__ __launch_bounds__(256)
void wmma_gemm_kernel(const float* __restrict__ A, const _Float16* __restrict__ Wt,
                      const float* __restrict__ bias, float* __restrict__ C,
                      int Nout) {
    __shared__ __align__(16) _Float16 As[16 * 32];
    const int tid  = threadIdx.x;
    const int wave = tid >> 5;
    const int lane = tid & 31;
    const int row0 = blockIdx.x * 16;
    const int n0   = blockIdx.y * 128 + wave * 16;
    const int lh   = lane & 15;
    const int hi   = lane >> 4;      // 0: lanes 0-15, 1: lanes 16-31

    // staging indices (threads 0..127 each move one float4 -> 4 halfs)
    const int sr  = tid >> 3;          // 0..15 (row)
    const int sc4 = (tid & 7) << 2;    // 0,4,..,28 (col)
    const float* aptr = A + (size_t)(row0 + sr) * KPAD + sc4;
    const _Float16* bptr = Wt + (size_t)(n0 + lh) * KPAD + hi * 8;

    v8f c = {};
    #pragma unroll
    for (int kt = 0; kt < KPAD / 32; ++kt) {
        const int k0 = kt * 32;
        if (kt) __syncthreads();
        if (tid < 128) {
            float4 v = *(const float4*)(aptr + k0);
            v4h h4;
            h4[0] = (_Float16)v.x; h4[1] = (_Float16)v.y;
            h4[2] = (_Float16)v.z; h4[3] = (_Float16)v.w;
            *(v4h*)&As[sr * 32 + sc4] = h4;
        }
        __syncthreads();
        v16h a, b;
        {
            v8h lo = *(const v8h*)&As[lh * 32 + hi * 8];
            v8h hh = *(const v8h*)&As[lh * 32 + hi * 8 + 16];
            #pragma unroll
            for (int i = 0; i < 8; i++) { a[i] = lo[i]; a[8 + i] = hh[i]; }
        }
        {
            v8h lo = *(const v8h*)(bptr + k0);
            v8h hh = *(const v8h*)(bptr + k0 + 16);
            #pragma unroll
            for (int i = 0; i < 8; i++) { b[i] = lo[i]; b[8 + i] = hh[i]; }
        }
        c = __builtin_amdgcn_wmma_f32_16x16x32_f16(false, a, false, b, (short)0, c, false, false);
    }
    int col = n0 + lh;
    float bv = bias ? bias[col] : 0.0f;
    #pragma unroll
    for (int r = 0; r < 8; r++) {
        int row = row0 + r + hi * 8;
        C[(size_t)row * Nout + col] = c[r] + bv;
    }
}

// ---------------- fused LayerNorm (+bias,+add) (+GELU) (+residual accumulate) ----------------
// one wave per row; 8 rows per block
__global__ __launch_bounds__(256)
void ln_fuse_kernel(const float* __restrict__ src, const float* __restrict__ add,
                    const float* __restrict__ bias,
                    const float* __restrict__ g, const float* __restrict__ b,
                    float* __restrict__ dst, int rows, int cols, int do_gelu, int accumulate) {
    int row = blockIdx.x * 8 + (threadIdx.x >> 5);
    if (row >= rows) return;
    int lane = threadIdx.x & 31;
    int per = cols >> 5;            // 8 (256) or 16 (512)
    float vals[16];
    const float* p = src + (size_t)row * cols;
    float s = 0.0f;
    for (int i = 0; i < per; i++) {
        int c = lane + (i << 5);
        float v = p[c];
        if (add)  v += add[(size_t)row * cols + c];
        if (bias) v += bias[c];
        vals[i] = v; s += v;
    }
    for (int o = 16; o > 0; o >>= 1) s += __shfl_xor(s, o, 32);
    float mean = s / (float)cols;
    float var = 0.0f;
    for (int i = 0; i < per; i++) { float d = vals[i] - mean; var += d * d; }
    for (int o = 16; o > 0; o >>= 1) var += __shfl_xor(var, o, 32);
    float inv = rsqrtf(var / (float)cols + 1e-5f);
    for (int i = 0; i < per; i++) {
        int c = lane + (i << 5);
        float y = (vals[i] - mean) * inv * g[c] + b[c];
        if (do_gelu) y = gelu_exact(y);
        if (accumulate) dst[(size_t)row * cols + c] += y;
        else            dst[(size_t)row * cols + c] = y;
    }
}

// ---------------- GATv2 edge score: e=lrelu(xl[s]+xr[d]); score[h]=dot(e_h, att_h); seg-max ----------------
__global__ __launch_bounds__(256)
void gat_score_kernel(const float* __restrict__ xl, const float* __restrict__ xr,
                      const int* __restrict__ src, const int* __restrict__ dst,
                      const float* __restrict__ att, float* __restrict__ score,
                      unsigned* __restrict__ nodemax, int E, int Etot) {
    int e = blockIdx.x * 8 + (threadIdx.x >> 5);
    if (e >= Etot) return;
    int lane = threadIdx.x & 31;
    int s = (e < E) ? src[e] : (e - E);
    int d = (e < E) ? dst[e] : (e - E);
    int h = lane >> 2, sub = lane & 3;
    int cbase = h * 32 + sub * 8;
    const float* ps = xl + (size_t)s * HID + cbase;
    const float* pd = xr + (size_t)d * HID + cbase;
    float part = 0.0f;
    #pragma unroll
    for (int j = 0; j < 8; j++) {
        float v = ps[j] + pd[j];
        v = (v > 0.0f) ? v : 0.2f * v;
        part += v * att[cbase + j];
    }
    part += __shfl_xor(part, 1, 32);
    part += __shfl_xor(part, 2, 32);
    if (sub == 0) {
        score[(size_t)e * NHEAD + h] = part;
        atomicMax(&nodemax[(size_t)d * NHEAD + h], f2ord(part));
    }
}

// ---------------- Transformer edge score: dot(q[d],k[s])/sqrt(D); seg-max ----------------
__global__ __launch_bounds__(256)
void tr_score_kernel(const float* __restrict__ q, const float* __restrict__ k,
                     const int* __restrict__ src, const int* __restrict__ dst,
                     float* __restrict__ score, unsigned* __restrict__ nodemax, int E) {
    int e = blockIdx.x * 8 + (threadIdx.x >> 5);
    if (e >= E) return;
    int lane = threadIdx.x & 31;
    int s = src[e], d = dst[e];
    int h = lane >> 2, sub = lane & 3;
    int cbase = h * 32 + sub * 8;
    const float* pq = q + (size_t)d * HID + cbase;
    const float* pk = k + (size_t)s * HID + cbase;
    float part = 0.0f;
    #pragma unroll
    for (int j = 0; j < 8; j++) part += pq[j] * pk[j];
    part += __shfl_xor(part, 1, 32);
    part += __shfl_xor(part, 2, 32);
    if (sub == 0) {
        part *= 0.17677669529663689f;     // 1/sqrt(32)
        score[(size_t)e * NHEAD + h] = part;
        atomicMax(&nodemax[(size_t)d * NHEAD + h], f2ord(part));
    }
}

// ---------------- pass2: ex = exp(score - max[d]); seg-sum ----------------
__global__ __launch_bounds__(256)
void edge_exp_kernel(float* __restrict__ score, const unsigned* __restrict__ nodemax,
                     float* __restrict__ nodesum, const int* __restrict__ dst,
                     int E, int Etot) {
    long long t = (long long)blockIdx.x * 256 + threadIdx.x;
    if (t >= (long long)Etot * NHEAD) return;
    int e = (int)(t >> 3), h = (int)(t & 7);
    int d = (e < E) ? dst[e] : (e - E);
    unsigned u = nodemax[(size_t)d * NHEAD + h];
    float m = (u == ORD_NEG_INF) ? 0.0f : ord2f(u);
    float ex = expf(score[t] - m);
    score[t] = ex;
    atomicAdd(&nodesum[(size_t)d * NHEAD + h], ex);
}

// ---------------- pass3: agg[d] += msg[s] * alpha ----------------
__global__ __launch_bounds__(256)
void edge_scatter_kernel(const float* __restrict__ msg, const float* __restrict__ score_ex,
                         const float* __restrict__ nodesum, const int* __restrict__ src,
                         const int* __restrict__ dst, float* __restrict__ agg,
                         int E, int Etot) {
    int e = blockIdx.x * 8 + (threadIdx.x >> 5);
    if (e >= Etot) return;
    int lane = threadIdx.x & 31;
    int s = (e < E) ? src[e] : (e - E);
    int d = (e < E) ? dst[e] : (e - E);
    int h = lane >> 2;
    int cbase = lane * 8;
    float alpha = score_ex[(size_t)e * NHEAD + h] /
                  (nodesum[(size_t)d * NHEAD + h] + 1e-16f);
    const float* mp = msg + (size_t)s * HID + cbase;
    float* ap = agg + (size_t)d * HID + cbase;
    #pragma unroll
    for (int j = 0; j < 8; j++) atomicAdd(&ap[j], mp[j] * alpha);
}

// ---------------- pooling ----------------
__global__ __launch_bounds__(256)
void pool_kernel(const float* __restrict__ h, const int* __restrict__ batch,
                 float* __restrict__ hsum, unsigned* __restrict__ hmaxo,
                 float* __restrict__ cnt, int N) {
    int n = blockIdx.x * 8 + (threadIdx.x >> 5);
    if (n >= N) return;
    int lane = threadIdx.x & 31;
    int b = batch[n];
    const float* p = h + (size_t)n * HID;
    #pragma unroll
    for (int i = 0; i < 8; i++) {
        int c = lane + (i << 5);
        float v = p[c];
        atomicAdd(&hsum[(size_t)b * HID + c], v);
        atomicMax(&hmaxo[(size_t)b * HID + c], f2ord(v));
    }
    if (lane == 0) atomicAdd(&cnt[b], 1.0f);
}

__global__ __launch_bounds__(256)
void ge_kernel(const float* __restrict__ hsum, const unsigned* __restrict__ hmaxo,
               const float* __restrict__ cnt, float* __restrict__ ge, int G) {
    int g = blockIdx.x * 8 + (threadIdx.x >> 5);
    if (g >= G) return;
    int lane = threadIdx.x & 31;
    float inv = 1.0f / fmaxf(cnt[g], 1.0f);
    #pragma unroll
    for (int i = 0; i < 8; i++) {
        int c = lane + (i << 5);
        ge[(size_t)g * 512 + c] = hsum[(size_t)g * HID + c] * inv;
        unsigned u = hmaxo[(size_t)g * HID + c];
        ge[(size_t)g * 512 + 256 + c] = (u == ORD_NEG_INF) ? 0.0f : ord2f(u);
    }
}

// ---------------- host launch ----------------
extern "C" void kernel_launch(void* const* d_in, const int* in_sizes, int n_in,
                              void* d_out, int out_size, void* d_ws, size_t ws_size,
                              hipStream_t stream) {
    const int N = in_sizes[0] / FIN;
    const int E = in_sizes[1] / 2;
    const int G = out_size / 512;
    const int Etot = E + N;                    // GAT self loops

    const float* x       = (const float*)d_in[0];
    const int*   ei      = (const int*)d_in[1];
    const int*   batch   = (const int*)d_in[2];
    const float* in_W    = (const float*)d_in[3];
    const float* in_b    = (const float*)d_in[4];
    const float* in_ln_g = (const float*)d_in[5];
    const float* in_ln_b = (const float*)d_in[6];
    const float* gat_Wl  = (const float*)d_in[7];
    const float* gat_bl  = (const float*)d_in[8];
    const float* gat_Wr  = (const float*)d_in[9];
    const float* gat_br  = (const float*)d_in[10];
    const float* gat_att = (const float*)d_in[11];
    const float* gat_bias= (const float*)d_in[12];
    const float* gat_ln_g= (const float*)d_in[13];
    const float* gat_ln_b= (const float*)d_in[14];
    const float* tr_Wq   = (const float*)d_in[15];
    const float* tr_bq   = (const float*)d_in[16];
    const float* tr_Wk   = (const float*)d_in[17];
    const float* tr_bk   = (const float*)d_in[18];
    const float* tr_Wv   = (const float*)d_in[19];
    const float* tr_bv   = (const float*)d_in[20];
    const float* tr_Ws   = (const float*)d_in[21];
    const float* tr_bs   = (const float*)d_in[22];
    const float* tr_ln_g = (const float*)d_in[23];
    const float* tr_ln_b = (const float*)d_in[24];
    const float* out_W   = (const float*)d_in[25];
    const float* out_b   = (const float*)d_in[26];
    const float* out_ln_g= (const float*)d_in[27];
    const float* out_ln_b= (const float*)d_in[28];

    const int* src = ei;
    const int* dst = ei + E;

    // workspace partition
    char* wsb = (char*)d_ws;
    size_t off = 0;
    auto carve = [&](size_t bytes) -> void* {
        void* p = wsb + off;
        off += (bytes + 255) & ~(size_t)255;
        return p;
    };
    float*    h       = (float*)carve((size_t)N * HID * 4);
    float*    xl      = (float*)carve((size_t)N * HID * 4);
    float*    xr      = (float*)carve((size_t)N * HID * 4);
    float*    agg     = (float*)carve((size_t)N * HID * 4);
    float*    xpad    = (float*)carve((size_t)N * 64 * 4);
    float*    score   = (float*)carve((size_t)Etot * NHEAD * 4);
    unsigned* nodemax = (unsigned*)carve((size_t)N * NHEAD * 4);
    float*    nodesum = (float*)carve((size_t)N * NHEAD * 4);
    float*    hsum    = (float*)carve((size_t)G * HID * 4);
    unsigned* hmaxo   = (unsigned*)carve((size_t)G * HID * 4);
    float*    cnt     = (float*)carve((size_t)G * 4);
    float*    ge      = (float*)carve((size_t)G * 512 * 4);
    float*    tmp512  = (float*)carve((size_t)G * 512 * 4);
    _Float16* in_Wt   = (_Float16*)carve((size_t)HID * 64 * 2);
    _Float16* gatWl_t = (_Float16*)carve((size_t)NLAYER * HID * HID * 2);
    _Float16* gatWr_t = (_Float16*)carve((size_t)NLAYER * HID * HID * 2);
    _Float16* trWq_t  = (_Float16*)carve((size_t)HID * HID * 2);
    _Float16* trWk_t  = (_Float16*)carve((size_t)HID * HID * 2);
    _Float16* trWv_t  = (_Float16*)carve((size_t)HID * HID * 2);
    _Float16* trWs_t  = (_Float16*)carve((size_t)HID * HID * 2);
    _Float16* out_Wt  = (_Float16*)carve((size_t)512 * 512 * 2);
    (void)ws_size; (void)n_in;

    auto fill = [&](void* p, unsigned v, long long n) {
        fill_u32_kernel<<<(unsigned)((n + 255) / 256), 256, 0, stream>>>((unsigned*)p, v, n);
    };
    auto prep = [&](const float* W, _Float16* Wt, int K, int Nout, int Kpad) {
        int n = Nout * Kpad;
        prep_wt_kernel<<<(n + 255) / 256, 256, 0, stream>>>(W, Wt, K, Nout, Kpad);
    };
    auto gemm = [&](const float* A, const _Float16* Wt, const float* bias, float* C,
                    int M, int Kpad, int Nout) {
        dim3 grid(M / 16, Nout / 128);
        if (Kpad == 64)
            wmma_gemm_kernel<64><<<grid, 256, 0, stream>>>(A, Wt, bias, C, Nout);
        else if (Kpad == 256)
            wmma_gemm_kernel<256><<<grid, 256, 0, stream>>>(A, Wt, bias, C, Nout);
        else
            wmma_gemm_kernel<512><<<grid, 256, 0, stream>>>(A, Wt, bias, C, Nout);
    };
    auto ln = [&](const float* s, const float* add, const float* bias,
                  const float* g, const float* b, float* d, int rows, int cols,
                  int dog, int acc) {
        ln_fuse_kernel<<<(rows + 7) / 8, 256, 0, stream>>>(s, add, bias, g, b, d, rows, cols, dog, acc);
    };

    // ---- weight prep (f16, transposed, K zero-padded to mult of 32) ----
    prep(in_W, in_Wt, FIN, HID, 64);
    for (int l = 0; l < NLAYER; l++) {
        prep(gat_Wl + (size_t)l * HID * HID, gatWl_t + (size_t)l * HID * HID, HID, HID, HID);
        prep(gat_Wr + (size_t)l * HID * HID, gatWr_t + (size_t)l * HID * HID, HID, HID, HID);
    }
    prep(tr_Wq, trWq_t, HID, HID, HID);
    prep(tr_Wk, trWk_t, HID, HID, HID);
    prep(tr_Wv, trWv_t, HID, HID, HID);
    prep(tr_Ws, trWs_t, HID, HID, HID);
    prep(out_W, out_Wt, 512, 512, 512);

    // ---- pad x to [N,64] ----
    {
        long long n = (long long)N * 64;
        pad_x_kernel<<<(unsigned)((n + 255) / 256), 256, 0, stream>>>(x, xpad, N);
    }

    // ---- input embedding: h = gelu(ln(x @ in_W + in_b)) ----
    gemm(xpad, in_Wt, in_b, agg, N, 64, HID);
    ln(agg, nullptr, nullptr, in_ln_g, in_ln_b, h, N, HID, 1, 0);

    // ---- 5 GATv2 layers ----
    const int eb  = (Etot + 7) / 8;                       // blocks, 8 edges each
    const long long eh = (long long)Etot * NHEAD;
    for (int l = 0; l < NLAYER; l++) {
        gemm(h, gatWl_t + (size_t)l * HID * HID, gat_bl + l * HID, xl, N, HID, HID);
        gemm(h, gatWr_t + (size_t)l * HID * HID, gat_br + l * HID, xr, N, HID, HID);
        fill(nodemax, ORD_NEG_INF, (long long)N * NHEAD);
        fill(nodesum, 0u, (long long)N * NHEAD);
        gat_score_kernel<<<eb, 256, 0, stream>>>(xl, xr, src, dst,
                                                 gat_att + (size_t)l * NHEAD * HDIM,
                                                 score, nodemax, E, Etot);
        edge_exp_kernel<<<(unsigned)((eh + 255) / 256), 256, 0, stream>>>(score, nodemax, nodesum, dst, E, Etot);
        fill(agg, 0u, (long long)N * HID);
        edge_scatter_kernel<<<eb, 256, 0, stream>>>(xl, score, nodesum, src, dst, agg, E, Etot);
        // h += gelu(ln(agg + gat_bias))
        ln(agg, nullptr, gat_bias + l * HID, gat_ln_g + l * HID, gat_ln_b + l * HID, h, N, HID, 1, 1);
    }

    // ---- TransformerConv (no self loops) ----
    const int eb2 = (E + 7) / 8;
    const long long eh2 = (long long)E * NHEAD;
    gemm(h, trWq_t, tr_bq, xl, N, HID, HID);              // q
    gemm(h, trWk_t, tr_bk, xr, N, HID, HID);              // k
    fill(nodemax, ORD_NEG_INF, (long long)N * NHEAD);
    fill(nodesum, 0u, (long long)N * NHEAD);
    tr_score_kernel<<<eb2, 256, 0, stream>>>(xl, xr, src, dst, score, nodemax, E);
    edge_exp_kernel<<<(unsigned)((eh2 + 255) / 256), 256, 0, stream>>>(score, nodemax, nodesum, dst, E, E);
    gemm(h, trWv_t, tr_bv, xr, N, HID, HID);              // v (k dead)
    fill(agg, 0u, (long long)N * HID);
    edge_scatter_kernel<<<eb2, 256, 0, stream>>>(xr, score, nodesum, src, dst, agg, E, E);
    gemm(h, trWs_t, tr_bs, xl, N, HID, HID);              // skip (q dead)
    // h += ln(agg + skip)   (no gelu)
    ln(agg, xl, nullptr, tr_ln_g, tr_ln_b, h, N, HID, 0, 1);

    // ---- global mean + max pool ----
    fill(hsum, 0u, (long long)G * HID);
    fill(hmaxo, ORD_NEG_INF, (long long)G * HID);
    fill(cnt, 0u, (long long)G);
    pool_kernel<<<(N + 7) / 8, 256, 0, stream>>>(h, batch, hsum, hmaxo, cnt, N);
    ge_kernel<<<(G + 7) / 8, 256, 0, stream>>>(hsum, hmaxo, cnt, ge, G);

    // ---- output projection: gelu(ln(ge @ out_W + out_b)) ----
    gemm(ge, out_Wt, out_b, tmp512, G, 512, 512);
    ln(tmp512, nullptr, nullptr, out_ln_g, out_ln_b, (float*)d_out, G, 512, 1, 0);
}
